// DynamicMoEBlock_89953795048081
// MI455X (gfx1250) — matrix operand order repaired and checked by reading, not compiled, not run on hardware
//
#include <hip/hip_runtime.h>
#include <hip/hip_bf16.h>
#include <math.h>

typedef __bf16 bf16;
typedef __attribute__((ext_vector_type(16))) __bf16 v16bf;
typedef __attribute__((ext_vector_type(8)))  __bf16 v8bf;
typedef __attribute__((ext_vector_type(8)))  float  v8f;

#define T_TOK 8192
#define H_DIM 1024
#define F_DIM 4096
#define E_NUM 8
#define MT 128
#define NT 128
#define KT 32
#define CAP (2 * T_TOK + E_NUM * MT)   // 17408 padded routed rows

__device__ __forceinline__ bf16 f2bf(float f) {
  unsigned u = __builtin_bit_cast(unsigned, f);
  unsigned r = (u + 0x7FFFu + ((u >> 16) & 1u)) >> 16;
  unsigned short s = (unsigned short)r;
  return __builtin_bit_cast(bf16, s);
}

// pack two floats into one dword of bf16 pair: lo -> bits[15:0], hi -> bits[31:16]
__device__ __forceinline__ unsigned pack_bf16x2(float lo, float hi) {
  unsigned ul = __builtin_bit_cast(unsigned, lo);
  unsigned uh = __builtin_bit_cast(unsigned, hi);
  ul = (ul + 0x7FFFu + ((ul >> 16) & 1u)) >> 16;
  uh = (uh + 0x7FFFu + ((uh >> 16) & 1u)) & 0xFFFF0000u;
  return ul | uh;
}

// ---------------- fp32 -> bf16 conversion ----------------
__global__ void cvt_f32_bf16(const float* __restrict__ in, bf16* __restrict__ out, size_t n) {
  size_t i = blockIdx.x * (size_t)blockDim.x + threadIdx.x;
  size_t stride = gridDim.x * (size_t)blockDim.x;
  for (; i < n; i += stride) out[i] = f2bf(in[i]);
}

// ---------------- router: logits -> softmax -> top-2 ----------------
__global__ void router_kernel(const float* __restrict__ x, const float* __restrict__ gw,
                              float* __restrict__ topw, int* __restrict__ topid,
                              int* __restrict__ counts) {
  int lane = threadIdx.x & 31;
  int wid  = threadIdx.x >> 5;
  int t = blockIdx.x * 8 + wid;
  float acc[E_NUM];
#pragma unroll
  for (int e = 0; e < E_NUM; ++e) acc[e] = 0.f;
  const float* xr = x + (size_t)t * H_DIM;
  for (int h = lane; h < H_DIM; h += 32) {
    float xv = xr[h];
#pragma unroll
    for (int e = 0; e < E_NUM; ++e) acc[e] += xv * gw[e * H_DIM + h];
  }
#pragma unroll
  for (int e = 0; e < E_NUM; ++e)
    for (int off = 16; off >= 1; off >>= 1)
      acc[e] += __shfl_xor(acc[e], off, 32);
  if (lane == 0) {
    float mx = acc[0];
    for (int e = 1; e < E_NUM; ++e) mx = fmaxf(mx, acc[e]);
    float p[E_NUM];
    for (int e = 0; e < E_NUM; ++e) p[e] = expf(acc[e] - mx);
    int e0 = 0;
    for (int e = 1; e < E_NUM; ++e) if (p[e] > p[e0]) e0 = e;
    int e1 = (e0 == 0) ? 1 : 0;
    for (int e = 0; e < E_NUM; ++e) if (e != e0 && p[e] > p[e1]) e1 = e;
    float w0 = p[e0], w1 = p[e1], inv = 1.f / (w0 + w1);
    topw[t * 2]     = w0 * inv;
    topw[t * 2 + 1] = w1 * inv;
    topid[t * 2]     = e0;
    topid[t * 2 + 1] = e1;
    atomicAdd(&counts[e0], 1);
    atomicAdd(&counts[e1], 1);
  }
}

// ---------------- tile-aligned per-expert segment offsets ----------------
__global__ void seg_offsets_kernel(const int* __restrict__ counts, int* __restrict__ segoff) {
  if (threadIdx.x == 0 && blockIdx.x == 0) {
    int o = 0;
    segoff[0] = 0;
    for (int e = 0; e < E_NUM; ++e) {
      o = ((o + counts[e] + MT - 1) / MT) * MT;
      segoff[e + 1] = o;
    }
  }
}

// ---------------- routed-row bookkeeping ----------------
__global__ void build_rows_kernel(const int* __restrict__ topid, const int* __restrict__ segoff,
                                  int* __restrict__ cursors, int* __restrict__ row_token,
                                  int* __restrict__ slot_row) {
  int t = blockIdx.x * blockDim.x + threadIdx.x;
  if (t >= T_TOK) return;
#pragma unroll
  for (int k = 0; k < 2; ++k) {
    int e = topid[t * 2 + k];
    int r = segoff[e] + atomicAdd(&cursors[e], 1);
    row_token[r] = t;
    slot_row[t * 2 + k] = r;
  }
}

// ---------------- unified WMMA GEMM: Out = act(A_bf16 @ W_f32 + bias) ----------------
// Tile 128x128, K-step 32; 8 waves each own a 32x64 subtile = 2x4 wmma_f32_16x16x32_bf16.
// A tile staged via gfx1250 async-to-LDS (bf16 direct); B tile staged fp32->bf16 into a
// column-major [n][k] LDS layout (padded stride 40 elems = 80B) so fragment reads are
// 2x ds_load_b128 per 16x16 tile. Next B tile is prefetched into registers during compute.
template <int KDIM, int NDIM, bool GELU>
__global__ __launch_bounds__(256)
void moe_gemm_wmma(const bf16* __restrict__ A, const float* __restrict__ W,
                   const float* __restrict__ bias,
                   const int* __restrict__ row_token, const int* __restrict__ segoff,
                   void* __restrict__ Out) {
  constexpr int BKS = 40;                // padded k-stride (elems) for Bs
  __shared__ bf16 As[MT * KT];           // [m][k], row stride 32 elems (64B)
  __shared__ bf16 Bs[NT * BKS];          // [n][k], col stride 40 elems (80B, 16B-aligned)

  const int tid = threadIdx.x;
  const int base_row = blockIdx.y * MT;
  const int n_base = blockIdx.x * NT;

  int e = 0;
  if (segoff) {
    if (base_row >= segoff[E_NUM]) return;
    while (e < E_NUM - 1 && base_row >= segoff[e + 1]) ++e;
  }
  const float* Wb = W + (size_t)e * KDIM * NDIM;
  const float* bb = bias + (size_t)e * NDIM;

  const int lane = tid & 31, wid = tid >> 5;
  const int sub_m = (wid & 3) * 32;
  const int sub_n = (wid >> 2) * 64;

  v8f acc[2][4];
#pragma unroll
  for (int mi = 0; mi < 2; ++mi)
#pragma unroll
    for (int ni = 0; ni < 4; ++ni)
      acc[mi][ni] = (v8f){0.f, 0.f, 0.f, 0.f, 0.f, 0.f, 0.f, 0.f};

  // ---- A staging coords: thread -> (row ar, 16-elem half ah) ----
  const int ar = tid >> 1, ah = (tid & 1) * 16;
  const int tokA = row_token ? row_token[base_row + ar] : (base_row + ar);
  const unsigned ldsA = (unsigned)(uintptr_t)&As[ar * KT + ah];
  if (tokA < 0) {  // padding row: zero once, never overwritten afterwards
    uint4 z = {0u, 0u, 0u, 0u};
    *((uint4*)&As[ar * KT + ah])     = z;
    *((uint4*)&As[ar * KT + ah] + 1) = z;
  }

  // ---- B staging coords: thread -> k rows {kp,kp+1}, cols n0..n0+7 ----
  const int kp = (tid >> 4) * 2;
  const int n0 = (tid & 15) * 8;
  const float4* wrow0 = (const float4*)(Wb + (size_t)kp * NDIM + n_base + n0);
  const float4* wrow1 = (const float4*)(Wb + (size_t)(kp + 1) * NDIM + n_base + n0);
  float4 r00 = wrow0[0], r01 = wrow0[1];
  float4 r10 = wrow1[0], r11 = wrow1[1];

  for (int k0 = 0; k0 < KDIM; k0 += KT) {
    __syncthreads();  // previous iteration's fragment reads done; LDS reusable

    // async A tile load (bf16 direct from global into LDS), EXEC-masked per lane
    if (tokA >= 0) {
      unsigned long long ga =
          (unsigned long long)(uintptr_t)(A + (size_t)tokA * KDIM + k0 + ah);
      asm volatile("global_load_async_to_lds_b128 %0, %1, off"
                   :: "v"(ldsA), "v"(ga) : "memory");
      asm volatile("global_load_async_to_lds_b128 %0, %1, off offset:16"
                   :: "v"(ldsA), "v"(ga) : "memory");
    }

    // B tile: convert current registers, pack k-pairs, store column-major
    {
      float a0[8] = {r00.x, r00.y, r00.z, r00.w, r01.x, r01.y, r01.z, r01.w};
      float a1[8] = {r10.x, r10.y, r10.z, r10.w, r11.x, r11.y, r11.z, r11.w};
#pragma unroll
      for (int j = 0; j < 8; ++j) {
        unsigned u = pack_bf16x2(a0[j], a1[j]);  // (k=kp, k=kp+1)
        *(unsigned*)&Bs[(n0 + j) * BKS + kp] = u;
      }
    }

    asm volatile("s_wait_asynccnt 0x0" ::: "memory");
    __syncthreads();  // staged tiles visible to all waves

    // prefetch next B tile into registers (overlaps with WMMA compute)
    if (k0 + KT < KDIM) {
      wrow0 += (KT * NDIM) / 4;
      wrow1 += (KT * NDIM) / 4;
      r00 = wrow0[0]; r01 = wrow0[1];
      r10 = wrow1[0]; r11 = wrow1[1];
    }

    // ---- fragments (all b128 LDS reads) ----
    v16bf afrag[2];
#pragma unroll
    for (int mi = 0; mi < 2; ++mi) {
      int m = sub_m + mi * 16 + (lane & 15);
      int kh = (lane >> 4) * 8;  // lanes 0-15: K 0..7/16..23 ; lanes 16-31: K 8..15/24..31
      v8bf lo = *(const v8bf*)&As[m * KT + kh];
      v8bf hi = *(const v8bf*)&As[m * KT + 16 + kh];
#pragma unroll
      for (int j = 0; j < 8; ++j) { afrag[mi][j] = lo[j]; afrag[mi][j + 8] = hi[j]; }
    }
    v16bf bfrag[4];
#pragma unroll
    for (int ni = 0; ni < 4; ++ni) {
      int n = sub_n + ni * 16 + (lane & 15);
      int kb = (lane >> 4) * 8;  // lanes 0-15: K 0..7/16..23 ; lanes 16-31: K 8..15/24..31
      v8bf lo = *(const v8bf*)&Bs[n * BKS + kb];
      v8bf hi = *(const v8bf*)&Bs[n * BKS + 16 + kb];
#pragma unroll
      for (int j = 0; j < 8; ++j) { bfrag[ni][j] = lo[j]; bfrag[ni][j + 8] = hi[j]; }
    }

#pragma unroll
    for (int mi = 0; mi < 2; ++mi)
#pragma unroll
      for (int ni = 0; ni < 4; ++ni)
        acc[mi][ni] = __builtin_amdgcn_wmma_f32_16x16x32_bf16(
            false, afrag[mi], false, bfrag[ni], (short)0, acc[mi][ni], false, false);
  }

  // ---- epilogue: bias (+ exact GELU) ----
#pragma unroll
  for (int mi = 0; mi < 2; ++mi) {
#pragma unroll
    for (int ni = 0; ni < 4; ++ni) {
      int colg  = n_base + sub_n + ni * 16 + (lane & 15);
      int rbase = base_row + sub_m + mi * 16 + ((lane >> 4) << 3);
      float bv = bb[colg];
#pragma unroll
      for (int r = 0; r < 8; ++r) {
        float v = acc[mi][ni][r] + bv;
        size_t off = (size_t)(rbase + r) * NDIM + colg;
        if (GELU) {
          v = 0.5f * v * (1.f + erff(v * 0.70710678118f));
          ((bf16*)Out)[off] = f2bf(v);
        } else {
          ((float*)Out)[off] = v;
        }
      }
    }
  }
}

// ---------------- combine: out += w0*y[row0] + w1*y[row1] ----------------
__global__ void combine_kernel(const float* __restrict__ y, const float* __restrict__ topw,
                               const int* __restrict__ slot_row, float* __restrict__ out) {
  size_t i = blockIdx.x * (size_t)blockDim.x + threadIdx.x;
  size_t total = (size_t)T_TOK * H_DIM;
  size_t stride = gridDim.x * (size_t)blockDim.x;
  for (; i < total; i += stride) {
    size_t t = i / H_DIM, h = i - t * H_DIM;
    int r0 = slot_row[t * 2], r1 = slot_row[t * 2 + 1];
    float v = out[i];
    v += topw[t * 2]     * y[(size_t)r0 * H_DIM + h];
    v += topw[t * 2 + 1] * y[(size_t)r1 * H_DIM + h];
    out[i] = v;
  }
}

extern "C" void kernel_launch(void* const* d_in, const int* in_sizes, int n_in,
                              void* d_out, int out_size, void* d_ws, size_t ws_size,
                              hipStream_t stream) {
  const float* x   = (const float*)d_in[0];
  const float* gw  = (const float*)d_in[1];
  const float* W1  = (const float*)d_in[2];
  const float* b1  = (const float*)d_in[3];
  const float* W2  = (const float*)d_in[4];
  const float* b2  = (const float*)d_in[5];
  const float* sW1 = (const float*)d_in[6];
  const float* sb1 = (const float*)d_in[7];
  const float* sW2 = (const float*)d_in[8];
  const float* sb2 = (const float*)d_in[9];
  float* out = (float*)d_out;

  char* ws = (char*)d_ws;
  size_t o = 0;
  auto alloc = [&](size_t bytes) -> void* {
    void* p = ws + o;
    o = (o + bytes + 255) & ~(size_t)255;
    return p;
  };
  bf16*  xb       = (bf16*)alloc((size_t)T_TOK * H_DIM * 2);   // 16.8 MB
  bf16*  hbuf     = (bf16*)alloc((size_t)CAP * F_DIM * 2);     // 142.6 MB (shared h, then routed h)
  float* ybuf     = (float*)alloc((size_t)CAP * H_DIM * 4);    // 71.3 MB
  float* topw     = (float*)alloc((size_t)T_TOK * 2 * 4);
  int*   topid    = (int*)alloc((size_t)T_TOK * 2 * 4);
  int*   slot_row = (int*)alloc((size_t)T_TOK * 2 * 4);
  int*   row_tok  = (int*)alloc((size_t)CAP * 4);
  int*   counts   = (int*)alloc(E_NUM * 4);
  int*   cursors  = (int*)alloc(E_NUM * 4);
  int*   segoff   = (int*)alloc((E_NUM + 1) * 4);

  hipMemsetAsync(counts, 0, E_NUM * 4, stream);
  hipMemsetAsync(cursors, 0, E_NUM * 4, stream);
  hipMemsetAsync(row_tok, 0xFF, (size_t)CAP * 4, stream);  // -1 = padding row

  cvt_f32_bf16<<<4096, 256, 0, stream>>>(x, xb, (size_t)T_TOK * H_DIM);
  router_kernel<<<T_TOK / 8, 256, 0, stream>>>(x, gw, topw, topid, counts);
  seg_offsets_kernel<<<1, 1, 0, stream>>>(counts, segoff);
  build_rows_kernel<<<T_TOK / 256, 256, 0, stream>>>(topid, segoff, cursors, row_tok, slot_row);

  // shared expert: out = gelu(x@sW1+sb1)@sW2 + sb2   (writes every element of d_out)
  moe_gemm_wmma<H_DIM, F_DIM, true><<<dim3(F_DIM / NT, T_TOK / MT), 256, 0, stream>>>(
      xb, sW1, sb1, nullptr, nullptr, hbuf);
  moe_gemm_wmma<F_DIM, H_DIM, false><<<dim3(H_DIM / NT, T_TOK / MT), 256, 0, stream>>>(
      hbuf, sW2, sb2, nullptr, nullptr, out);

  // routed experts over tile-aligned row segments
  moe_gemm_wmma<H_DIM, F_DIM, true><<<dim3(F_DIM / NT, CAP / MT), 256, 0, stream>>>(
      xb, W1, b1, row_tok, segoff, hbuf);
  moe_gemm_wmma<F_DIM, H_DIM, false><<<dim3(H_DIM / NT, CAP / MT), 256, 0, stream>>>(
      hbuf, W2, b2, nullptr, segoff, ybuf);

  combine_kernel<<<8192, 256, 0, stream>>>(ybuf, topw, slot_row, out);
}
